// MultiHeadedAttention_10703058502176
// MI455X (gfx1250) — matrix operand order repaired and checked by reading, compiled, not run on hardware
//
#include <hip/hip_runtime.h>
#include <cstdint>

// ---------------------------------------------------------------------------
// MI455X (gfx1250) multi-head attention forward.
// - All matrix math via v_wmma_f32_16x16x32_bf16 (wave32, f32 accum).
// - Attention K-chunk staged global->LDS by the Tensor Data Mover
//   (tensor_load_to_lds + s_wait_tensorcnt).
// Workspace (bf16): Q | K | V | X, each 8192x1024 (16 MB) -> 64 MB total.
// ---------------------------------------------------------------------------

typedef __bf16 bf16;
typedef __attribute__((ext_vector_type(16))) bf16  bf16x16;
typedef __attribute__((ext_vector_type(8)))  float v8f;
typedef __attribute__((ext_vector_type(4)))  unsigned int u32x4;
typedef __attribute__((ext_vector_type(4)))  int i32x4;
typedef __attribute__((ext_vector_type(8)))  int i32x8;

#define WMMA_BF16(a, b, c)                                                     \
  __builtin_amdgcn_wmma_f32_16x16x32_bf16(false, (a), false, (b), (short)0,    \
                                          (c), false, false)

__device__ __forceinline__ v8f vzero() {
  v8f z = {0.f, 0.f, 0.f, 0.f, 0.f, 0.f, 0.f, 0.f};
  return z;
}

// ---- Tensor Data Mover: 2D bf16 tile (rows x row_elems) global -> LDS -----
// D# layout per CDNA5 ISA ch.8: group0 = {count|flags, lds_addr,
// global_addr[31:0], global_addr[56:32]|type=2}; group1 packs data_size,
// tensor_dim0/1, tile_dim0/1/2, tensor_dim0_stride (fields not dword-aligned).
__device__ __forceinline__ void tdm_load_2d_to_lds(unsigned lds_off,
                                                   const void* gptr,
                                                   unsigned rows,
                                                   unsigned row_elems,
                                                   unsigned row_stride_elems) {
  const unsigned long long ga = (unsigned long long)(size_t)gptr;
  u32x4 g0;
  g0[0] = 1u;                                                // count=1, user D#
  g0[1] = lds_off;                                           // lds_addr (bytes)
  g0[2] = (unsigned)(ga & 0xFFFFFFFFull);                    // global_addr lo
  g0[3] = (unsigned)((ga >> 32) & 0x1FFFFFFull) | (2u << 30);// addr hi | type=2
  i32x8 g1;
  g1[0] = (int)(1u << 16);                                   // data_size=2B, mask=0
  g1[1] = (int)((row_elems & 0xFFFFu) << 16);                // tensor_dim0 lo16
  g1[2] = (int)(((row_elems >> 16) & 0xFFFFu) |
                ((rows & 0xFFFFu) << 16));                   // td0 hi | td1 lo
  g1[3] = (int)(((rows >> 16) & 0xFFFFu) |
                ((row_elems & 0xFFFFu) << 16));              // td1 hi | tile_dim0
  g1[4] = (int)(rows & 0xFFFFu);                             // tile_dim1 (dim2=0)
  g1[5] = (int)row_stride_elems;                             // dim0_stride lo32
  g1[6] = 0;                                                 // stride hi | d1s lo
  g1[7] = 0;
  const i32x4 z4 = {0, 0, 0, 0};
#if defined(__clang_major__) && (__clang_major__ >= 23)
  const i32x8 z8 = {0, 0, 0, 0, 0, 0, 0, 0};
  __builtin_amdgcn_tensor_load_to_lds(g0, g1, z4, z4, z8, 0);
#else
  __builtin_amdgcn_tensor_load_to_lds(g0, g1, z4, z4, 0);
#endif
}

// ---- fragment loaders -----------------------------------------------------
// A fragment (16x32, bf16): lane holds row (lane&15); elems 0..7 -> K = kh+j,
// elems 8..15 -> K = 16+kh+j, kh = (lane>>4)*8.  (ISA 7.12.2 table)
__device__ __forceinline__ bf16x16 frag_a(const bf16* p, int ld, int lane, int k0) {
  const bf16* r = p + (size_t)(lane & 15) * ld + k0 + ((lane >> 4) << 3);
  bf16x16 f;
#pragma unroll
  for (int j = 0; j < 8; ++j) { f[j] = r[j]; f[8 + j] = r[16 + j]; }
  return f;
}
__device__ __forceinline__ bf16x16 frag_a(const float* p, int ld, int lane, int k0) {
  const float* r = p + (size_t)(lane & 15) * ld + k0 + ((lane >> 4) << 3);
  bf16x16 f;
#pragma unroll
  for (int j = 0; j < 8; ++j) { f[j] = (bf16)r[j]; f[8 + j] = (bf16)r[16 + j]; }
  return f;
}
// B fragment (32x16, bf16): lane holds column (lane&15); elems 0..7 ->
// K = kh+j, elems 8..15 -> K = 16+kh+j, kh = (lane>>4)*8.  Source rows are contiguous.
__device__ __forceinline__ bf16x16 frag_b(const bf16* p, int ld, int lane, int k0) {
  const bf16* r = p + (size_t)(lane & 15) * ld + k0 + ((lane >> 4) << 3);
  bf16x16 f;
#pragma unroll
  for (int j = 0; j < 8; ++j) { f[j] = r[j]; f[8 + j] = r[16 + j]; }
  return f;
}
__device__ __forceinline__ bf16x16 frag_b(const float* p, int ld, int lane, int k0) {
  const float* r = p + (size_t)(lane & 15) * ld + k0 + ((lane >> 4) << 3);
  bf16x16 f;
#pragma unroll
  for (int j = 0; j < 8; ++j) { f[j] = (bf16)r[j]; f[8 + j] = (bf16)r[16 + j]; }
  return f;
}

// ---------------------------------------------------------------------------
// GEMM: C[M,N] = A[M,K] * W[N,K]^T + bias[N]   (torch Linear semantics)
// 256 threads = 8 waves, wave tile 64x64 (4x4 wmma accums), block 128x256.
// ---------------------------------------------------------------------------
template <typename TA, typename TO>
__global__ __launch_bounds__(256) void gemm_nt_bias(
    const TA* __restrict__ A, const float* __restrict__ W,
    const float* __restrict__ bias, TO* __restrict__ C, int M, int N, int K) {
  const int lane = threadIdx.x & 31;
  const int wave = threadIdx.x >> 5;
  const int row0 = blockIdx.y * 128 + (wave & 1) * 64;
  const int col0 = blockIdx.x * 256 + (wave >> 1) * 64;

  v8f acc[4][4];
#pragma unroll
  for (int i = 0; i < 4; ++i)
#pragma unroll
    for (int j = 0; j < 4; ++j) acc[i][j] = vzero();

  for (int k0 = 0; k0 < K; k0 += 32) {
    if (k0 + 32 < K) {  // hint next K tile into cache (global_prefetch_b8)
      __builtin_prefetch(A + (size_t)(row0 + (lane & 15)) * K + k0 + 32, 0, 1);
      __builtin_prefetch(W + (size_t)(col0 + (lane & 15)) * K + k0 + 32, 0, 1);
    }
    bf16x16 af[4], bw[4];
#pragma unroll
    for (int i = 0; i < 4; ++i)
      af[i] = frag_a(A + (size_t)(row0 + 16 * i) * K, K, lane, k0);
#pragma unroll
    for (int j = 0; j < 4; ++j)
      bw[j] = frag_b(W + (size_t)(col0 + 16 * j) * K, K, lane, k0);
#pragma unroll
    for (int i = 0; i < 4; ++i)
#pragma unroll
      for (int j = 0; j < 4; ++j) acc[i][j] = WMMA_BF16(af[i], bw[j], acc[i][j]);
  }

  const int nlo = lane & 15;
  const int mhi = (lane >> 4) * 8;  // C layout: m = r + 8*(lane>>4), n = lane&15
#pragma unroll
  for (int j = 0; j < 4; ++j) {
    const int n = col0 + 16 * j + nlo;
    const float bv = bias[n];
#pragma unroll
    for (int i = 0; i < 4; ++i) {
      const int mb = row0 + 16 * i + mhi;
#pragma unroll
      for (int r = 0; r < 8; ++r)
        C[(size_t)(mb + r) * N + n] = (TO)(acc[i][j][r] + bv);
    }
  }
}

// ---------------------------------------------------------------------------
// Attention: one (b, h, 128 query rows) per block. Q/K/V/X are [B*S, 1024]
// bf16 with column = h*64 + d. Streaming over keys; e' = exp(s/8)*policy+EPS/S
// so p = e'/rowsum(e') exactly reproduces (sc+EPS/S)/(sum sc + EPS).
// ---------------------------------------------------------------------------
__global__ __launch_bounds__(256) void attn_kernel(
    const bf16* __restrict__ Q, const bf16* __restrict__ Km,
    const bf16* __restrict__ Vm, const float* __restrict__ policy,
    bf16* __restrict__ X) {
  constexpr int S = 1024, D = 1024, DK = 64;
  __shared__ __align__(16) bf16 Ks[128][64];   // key-chunk (TDM target)
  __shared__ __align__(16) bf16 Vt[64][128];   // value-chunk, transposed (d x key)
  __shared__ __align__(16) bf16 Es[8][16][32]; // per-wave P tile (C->A relayout)

  const int lane = threadIdx.x & 31;
  const int wave = threadIdx.x >> 5;
  const int b = blockIdx.z, h = blockIdx.y;
  const int q0 = blockIdx.x * 128;
  const int nlo = lane & 15;
  const int mhi = (lane >> 4) * 8;

  const size_t rowQ = (size_t)(b * S + q0 + wave * 16);
  const bf16* qbase = Q + rowQ * D + h * DK;
  const bf16x16 qa0 = frag_a(qbase, D, lane, 0);
  const bf16x16 qa1 = frag_a(qbase, D, lane, 32);

  bf16x16 ones;
#pragma unroll
  for (int i = 0; i < 16; ++i) ones[i] = (bf16)1.0f;

  v8f xacc[4];
#pragma unroll
  for (int t = 0; t < 4; ++t) xacc[t] = vzero();
  v8f rs = vzero();

  for (int kc = 0; kc < S; kc += 128) {
    __syncthreads();  // protect previous chunk's LDS reads
    // K chunk (128 rows x 64 bf16) via Tensor Data Mover, issued by wave 0
    if (wave == 0) {
      tdm_load_2d_to_lds((unsigned)(size_t)&Ks[0][0],
                         Km + ((size_t)(b * S + kc)) * D + h * DK,
                         /*rows=*/128, /*row_elems=*/64,
                         /*row_stride_elems=*/(unsigned)D);
    }
    // stage V chunk transposed: Vt[d][key]
    for (int t = threadIdx.x; t < 128 * 64; t += 256) {
      const int key = t >> 6, d = t & 63;
      Vt[d][key] = Vm[((size_t)(b * S + kc + key)) * D + h * DK + d];
    }
    if (wave == 0) __builtin_amdgcn_s_wait_tensorcnt(0);
    __syncthreads();

    for (int j = 0; j < 128; j += 32) {
      // two 16-key score tiles -> exp -> per-wave LDS scratch (A layout src)
#pragma unroll
      for (int t2 = 0; t2 < 2; ++t2) {
        const int keyb = j + t2 * 16;
        const bf16x16 kb0 = frag_b(&Ks[keyb][0], 64, lane, 0);
        const bf16x16 kb1 = frag_b(&Ks[keyb][0], 64, lane, 32);
        v8f s = vzero();
        s = WMMA_BF16(qa0, kb0, s);
        s = WMMA_BF16(qa1, kb1, s);
        const float pol = policy[(size_t)b * S + kc + keyb + nlo];
#pragma unroll
        for (int r = 0; r < 8; ++r) {
          const float e = __expf(s[r] * 0.125f) * pol + 9.765625e-10f;
          Es[wave][mhi + r][t2 * 16 + nlo] = (bf16)e;
        }
      }
      // reload P as A fragment (same-wave LDS ops are in-order)
      const bf16x16 pf = frag_a(&Es[wave][0][0], 32, lane, 0);
      rs = WMMA_BF16(pf, ones, rs);  // row-sums, same C layout as xacc
#pragma unroll
      for (int t3 = 0; t3 < 4; ++t3) {
        const bf16x16 vb = frag_b(&Vt[t3 * 16][j], 128, lane, 0);
        xacc[t3] = WMMA_BF16(pf, vb, xacc[t3]);
      }
    }
  }

  // normalize and store X (bf16, [B*S, D] with col = h*64 + d)
#pragma unroll
  for (int t3 = 0; t3 < 4; ++t3) {
#pragma unroll
    for (int r = 0; r < 8; ++r) {
      const size_t row = rowQ + mhi + r;
      X[row * D + h * DK + t3 * 16 + nlo] = (bf16)(xacc[t3][r] / rs[r]);
    }
  }
}

// ---------------------------------------------------------------------------
extern "C" void kernel_launch(void* const* d_in, const int* in_sizes, int n_in,
                              void* d_out, int out_size, void* d_ws,
                              size_t ws_size, hipStream_t stream) {
  (void)in_sizes; (void)n_in; (void)out_size; (void)ws_size;
  const float* query  = (const float*)d_in[0];
  const float* key    = (const float*)d_in[1];
  const float* value  = (const float*)d_in[2];
  const float* policy = (const float*)d_in[3];
  const float* wq_w = (const float*)d_in[4];
  const float* wq_b = (const float*)d_in[5];
  const float* wk_w = (const float*)d_in[6];
  const float* wk_b = (const float*)d_in[7];
  const float* wv_w = (const float*)d_in[8];
  const float* wv_b = (const float*)d_in[9];
  const float* dw   = (const float*)d_in[10];
  const float* db   = (const float*)d_in[11];
  float* out = (float*)d_out;

  const int M = 8192, N = 1024, K = 1024;
  const size_t MAT = (size_t)M * N;
  bf16* Qw = (bf16*)d_ws;
  bf16* Kw = Qw + MAT;
  bf16* Vw = Kw + MAT;
  bf16* Xw = Vw + MAT;

  const dim3 gg(N / 256, M / 128), bb(256);
  gemm_nt_bias<float, bf16><<<gg, bb, 0, stream>>>(query, wq_w, wq_b, Qw, M, N, K);
  gemm_nt_bias<float, bf16><<<gg, bb, 0, stream>>>(key,   wk_w, wk_b, Kw, M, N, K);
  gemm_nt_bias<float, bf16><<<gg, bb, 0, stream>>>(value, wv_w, wv_b, Vw, M, N, K);
  attn_kernel<<<dim3(8, 16, 8), bb, 0, stream>>>(Qw, Kw, Vw, policy, Xw);
  gemm_nt_bias<bf16, float><<<gg, bb, 0, stream>>>(Xw, dw, db, out, M, N, K);
}
